// PFGNN_Net_37598143709459
// MI455X (gfx1250) — compile-verified, matrix operand order checked by reading, not run on hardware
//
#include <hip/hip_runtime.h>

// ---------------------------------------------------------------------------
// PFGNN on MI455X (gfx1250): bf16 WMMA GEMMs (v_wmma_f32_16x16x32_bf16) with
// LDS-staged weights + register-resident A fragments, fp32 scatter/aggregate
// with native f32 atomics, particle-shared edge GEMM.
//
// Assumed d_in leaf order (insertion-order flatten of setup_inputs()):
//   0 node_emb(20480x128 f32)  1 edge_emb(163840x128 f32)
//   conv0 layers l=0..4: [2+4l]=We [3+4l]=be [4+4l]=W [5+4l]=b ; 22=Wr 23=br
//   conv1: 24=We 25=be 26=W 27=b 28=Wr 29=br ; conv2: 30..35
//   policy: 36=W1 37=b1 38=W2 39=b2 40=W3(128x1) 41=b3
//   relabel: 42=W1 43=b1 44=W2 45=b2
//   mlp: 46=W1(384x256) 47=b1 48=W2(256x128) 49=b2 50=W3(128x10) 51=b3
//   52 edge_index(2x163840 i32, row0=src row1=dst)  53 batch(unused; b=v/80)
// d_out: [0..2559] out(256x10 f32) | [2560..4607] log_probs(256x8) | [4608]=256
// ---------------------------------------------------------------------------

typedef __attribute__((ext_vector_type(16))) __bf16 v16bf;
typedef __attribute__((ext_vector_type(8)))  float  v8f;

constexpr int DIMC  = 128;
constexpr int NN    = 20480;    // nodes
constexpr int NE    = 163840;   // edges
constexpr int PP    = 4;        // particles
constexpr int BB    = 256;      // graphs
constexpr int PER   = 80;       // nodes per graph
constexpr int NCOMP = PP * NN;  // 81920
constexpr int GG    = PP * BB;  // 1024

__device__ __forceinline__ unsigned short f2bf(float f) {
  unsigned int u = __float_as_uint(f);
  u += 0x7fffu + ((u >> 16) & 1u);            // round-to-nearest-even
  return (unsigned short)(u >> 16);
}
__device__ __forceinline__ float bf2f(unsigned short h) {
  return __uint_as_float(((unsigned int)h) << 16);
}
__device__ __forceinline__ float rng01(unsigned long long x) {
  x += 0x9E3779B97F4A7C15ull;
  x = (x ^ (x >> 30)) * 0xBF58476D1CE4E5B9ull;
  x = (x ^ (x >> 27)) * 0x94D049BB133111EBull;
  x ^= x >> 31;
  return (float)((x >> 40) & 0xFFFFFFull) * (1.0f / 16777216.0f);
}

// ---- weight convert: W (Kin x Nout, f32, row-major) -> Wt (Npad x Kin, bf16)
__global__ void cvt_weight_t_k(const float* __restrict__ W,
                               unsigned short* __restrict__ Wt,
                               int Kin, int Nout, int Npad) {
  size_t total = (size_t)Npad * Kin;
  for (size_t i = blockIdx.x * (size_t)blockDim.x + threadIdx.x; i < total;
       i += (size_t)gridDim.x * blockDim.x) {
    int n = (int)(i / Kin), k = (int)(i % Kin);
    Wt[i] = (n < Nout) ? f2bf(W[(size_t)k * Nout + n]) : (unsigned short)0;
  }
}

__global__ void f32_to_bf16_k(const float* __restrict__ a,
                              unsigned short* __restrict__ o, size_t cnt) {
  for (size_t i = blockIdx.x * (size_t)blockDim.x + threadIdx.x; i < cnt;
       i += (size_t)gridDim.x * blockDim.x)
    o[i] = f2bf(a[i]);
}

__global__ void fill_f32_k(float* __restrict__ p, float v, size_t cnt) {
  for (size_t i = blockIdx.x * (size_t)blockDim.x + threadIdx.x; i < cnt;
       i += (size_t)gridDim.x * blockDim.x)
    p[i] = v;
}

__global__ void add2_bf16_k(const float* __restrict__ a,
                            const float* __restrict__ b,
                            unsigned short* __restrict__ o, size_t cnt) {
  for (size_t i = blockIdx.x * (size_t)blockDim.x + threadIdx.x; i < cnt;
       i += (size_t)gridDim.x * blockDim.x)
    o[i] = f2bf(a[i] + b[i]);
}

// ---------------------------------------------------------------------------
// Core WMMA GEMM: D[M x Ntrue] = act(A[M x K]_bf16 * Wt[Npad x K]^T + bias)
// K = KSTEPS*32 (compile-time). 8 waves/block, each wave = one 16-row strip.
// Wt staged in LDS once per block (ds_load_b128 fragments thereafter);
// all A fragments for the strip preloaded into registers once.
// ---------------------------------------------------------------------------
template <int KSTEPS>
__global__ __launch_bounds__(256) void gemm_wmma_t(
    const unsigned short* __restrict__ A, const unsigned short* __restrict__ Wt,
    const float* __restrict__ bias, float* __restrict__ outF,
    unsigned short* __restrict__ outB, int M, int Ntrue, int Npad, int relu) {
  constexpr int K = KSTEPS * 32;
  extern __shared__ unsigned short lds[];

  // cooperative stage of Wt (Npad x K bf16) into LDS, b128 granularity
  {
    const uint4* sv = (const uint4*)Wt;
    uint4* dv = (uint4*)lds;
    const int total = (Npad * K) >> 3;  // uint4 count
    for (int i = threadIdx.x; i < total; i += 256) dv[i] = sv[i];
  }
  __syncthreads();   // all M launches are exact multiples of 128 rows

  const int tid = threadIdx.x;
  const int wave = tid >> 5;
  const int lane = tid & 31;
  const int half = lane >> 4;   // 0: lanes 0-15, 1: lanes 16-31
  const int l16  = lane & 15;
  const int half4 = half * 4;
  const int rowbase = (blockIdx.x * 8 + wave) * 16;
  if (rowbase >= M) return;     // whole-wave exit: EXEC all-ones for WMMA
  int rA = rowbase + l16;
  if (rA >= M) rA = M - 1;

  union U { v16bf v; uint4 q[2]; };

  // preload all A fragments for this wave's 16-row strip (registers)
  U fa[KSTEPS];
  {
    const unsigned int* ap = (const unsigned int*)(A + (size_t)rA * K);
#pragma unroll
    for (int k = 0; k < KSTEPS; ++k) {
      const unsigned int* p = ap + k * 16;          // 32 bf16 per k-step
      fa[k].q[0] = *(const uint4*)(p + half4);      // K = half*8 + 0..7
      fa[k].q[1] = *(const uint4*)(p + half4 + 8);  // K = 16 + half*8 + 0..7
    }
  }

  const int ntiles = Npad >> 4;
  for (int nt = 0; nt < ntiles; ++nt) {
    const int nB = nt * 16 + l16;                   // B column for this lane
    const unsigned int* bp = (const unsigned int*)(lds + (size_t)nB * K);
    v8f acc = {0.f, 0.f, 0.f, 0.f, 0.f, 0.f, 0.f, 0.f};
    if constexpr (KSTEPS <= 8) {
      // load ALL B fragments (ds_load_b128 clause), then back-to-back WMMAs
      U fb[KSTEPS];
#pragma unroll
      for (int k = 0; k < KSTEPS; ++k) {
        const unsigned int* p = bp + k * 16;
        fb[k].q[0] = *(const uint4*)(p + half4);
        fb[k].q[1] = *(const uint4*)(p + half4 + 8);
      }
#pragma unroll
      for (int k = 0; k < KSTEPS; ++k)
        acc = __builtin_amdgcn_wmma_f32_16x16x32_bf16(
            false, fa[k].v, false, fb[k].v, (short)0, acc, false, false);
    } else {
      // large-K (tiny M) path: per-step B load + WMMA
#pragma unroll
      for (int k = 0; k < KSTEPS; ++k) {
        U fb;
        const unsigned int* p = bp + k * 16;
        fb.q[0] = *(const uint4*)(p + half4);
        fb.q[1] = *(const uint4*)(p + half4 + 8);
        acc = __builtin_amdgcn_wmma_f32_16x16x32_bf16(
            false, fa[k].v, false, fb.v, (short)0, acc, false, false);
      }
    }
    const int n = nt * 16 + l16;
    if (n < Ntrue) {
      const float bb = bias ? bias[n] : 0.f;
#pragma unroll
      for (int v = 0; v < 8; ++v) {
        const int row = rowbase + half * 8 + v;     // C/D layout
        if (row < M) {
          float x = acc[v] + bb;
          if (relu) x = fmaxf(x, 0.f);
          const size_t o = (size_t)row * Ntrue + n;
          if (outF) outF[o] = x;
          if (outB) outB[o] = f2bf(x);
        }
      }
    }
  }
}

// ---- msg = relu(h[src]+e); agg[dst] += msg   (one wave per edge, float4/lane)
__global__ __launch_bounds__(256) void edge_scatter_k(
    const float* __restrict__ h, const float* __restrict__ e,
    const int* __restrict__ src, const int* __restrict__ dst,
    float* __restrict__ agg, int E, int ET, int n) {
  const int wv = blockIdx.x * 8 + (threadIdx.x >> 5);
  if (wv >= ET) return;
  const int lane = threadIdx.x & 31;
  const int p = wv / E;
  const int el = wv - p * E;
  const int s = src[el] + p * n;
  const int d = dst[el] + p * n;
  const float4 hv = *(const float4*)(h + (size_t)s * DIMC + lane * 4);
  const float4 ev = *(const float4*)(e + (size_t)el * DIMC + lane * 4);
  float m0 = fmaxf(hv.x + ev.x, 0.f), m1 = fmaxf(hv.y + ev.y, 0.f);
  float m2 = fmaxf(hv.z + ev.z, 0.f), m3 = fmaxf(hv.w + ev.w, 0.f);
  float* ag = agg + (size_t)d * DIMC + lane * 4;
  __hip_atomic_fetch_add(ag + 0, m0, __ATOMIC_RELAXED, __HIP_MEMORY_SCOPE_AGENT);
  __hip_atomic_fetch_add(ag + 1, m1, __ATOMIC_RELAXED, __HIP_MEMORY_SCOPE_AGENT);
  __hip_atomic_fetch_add(ag + 2, m2, __ATOMIC_RELAXED, __HIP_MEMORY_SCOPE_AGENT);
  __hip_atomic_fetch_add(ag + 3, m3, __ATOMIC_RELAXED, __HIP_MEMORY_SCOPE_AGENT);
}

// ---- replicate conv0 output into P particle copies (f32 + bf16)
__global__ void replicate4_k(const float* __restrict__ hf,
                             const unsigned short* __restrict__ hb,
                             float* __restrict__ of,
                             unsigned short* __restrict__ ob, size_t cnt) {
  for (size_t i = blockIdx.x * (size_t)blockDim.x + threadIdx.x; i < cnt;
       i += (size_t)gridDim.x * blockDim.x) {
    const float v = hf[i];
    const unsigned short bv = hb[i];
#pragma unroll
    for (int p = 0; p < PP; ++p) {
      of[(size_t)p * cnt + i] = v;
      ob[(size_t)p * cnt + i] = bv;
    }
  }
}

// ---- s[row] = dot(X_bf16[row], w_f32) + b   (wave per row, shfl reduce)
__global__ __launch_bounds__(256) void rowdot_k(
    const unsigned short* __restrict__ X, const float* __restrict__ w,
    const float* __restrict__ b, float* __restrict__ s, int M) {
  const int row = blockIdx.x * 8 + (threadIdx.x >> 5);
  if (row >= M) return;
  const int lane = threadIdx.x & 31;
  const unsigned int* xp = (const unsigned int*)(X + (size_t)row * DIMC);
  float acc = 0.f;
#pragma unroll
  for (int q = 0; q < 2; ++q) {
    const unsigned int u = xp[lane * 2 + q];
    const int c = lane * 4 + q * 2;
    acc += bf2f((unsigned short)(u & 0xffff)) * w[c];
    acc += bf2f((unsigned short)(u >> 16)) * w[c + 1];
  }
  for (int off = 16; off; off >>= 1) acc += __shfl_down(acc, off);
  if (lane == 0) s[row] = acc + b[0];
}

// ---- per-graph softmax over PER nodes + categorical sample + log-prob out
__global__ void sample_policy_k(const float* __restrict__ s,
                                int* __restrict__ perm,
                                float* __restrict__ dlp, int depth) {
  const int g = blockIdx.x * blockDim.x + threadIdx.x;
  if (g >= GG) return;
  const int p = g >> 8, b = g & 255;
  const int base = p * NN + b * PER;
  float mx = -3.4e38f;
  for (int j = 0; j < PER; ++j) mx = fmaxf(mx, s[base + j]);
  float sum = 0.f;
  for (int j = 0; j < PER; ++j) sum += __expf(s[base + j] - mx);
  const float u = rng01(((unsigned long long)(2 * depth) << 40) ^
                        (unsigned long long)(g + 0x51ed));
  const float target = u * sum;
  float c = 0.f;
  int pick = PER - 1;
  for (int j = 0; j < PER; ++j) {
    c += __expf(s[base + j] - mx);
    if (c >= target) { pick = j; break; }
  }
  perm[g] = base + pick;
  const float prob = __expf(s[base + pick] - mx) / sum;
  dlp[(size_t)b * (2 * PP) + depth * PP + p] = __logf(prob + 1e-30f);
}

__global__ void gather_rows_bf_k(const float* __restrict__ nodes,
                                 const int* __restrict__ perm,
                                 unsigned short* __restrict__ sel) {
  const int id = blockIdx.x * blockDim.x + threadIdx.x;
  if (id >= GG * DIMC) return;
  const int g = id >> 7, c = id & 127;
  sel[id] = f2bf(nodes[(size_t)perm[g] * DIMC + c]);
}

__global__ void scatter_mul_k(float* __restrict__ nodes,
                              const int* __restrict__ perm,
                              const float* __restrict__ rel) {
  const int id = blockIdx.x * blockDim.x + threadIdx.x;
  if (id >= GG * DIMC) return;
  const int g = id >> 7, c = id & 127;
  nodes[(size_t)perm[g] * DIMC + c] *= rel[id];
}

// ---- particle-filter resampling weights
__global__ void resample_k(float* __restrict__ wlog, int* __restrict__ idx,
                           float* __restrict__ wexp, int depth) {
  const int b = blockIdx.x * blockDim.x + threadIdx.x;
  if (b >= BB) return;
  float wp[PP], ws[PP], wl[PP];
  float sum = 0.f;
#pragma unroll
  for (int p = 0; p < PP; ++p) {
    wp[p] = __expf(wlog[p * BB + b]);
    ws[p] = 0.5f * wp[p] + 0.125f;
    sum += ws[p];
  }
#pragma unroll
  for (int p = 0; p < PP; ++p) {
    const float u = rng01(((unsigned long long)(2 * depth + 1) << 40) ^
                          (unsigned long long)(p * BB + b + 0xBEEF)) * sum;
    int ip = PP - 1;
    float c = 0.f;
    for (int q = 0; q < PP; ++q) {
      c += ws[q];
      if (c >= u) { ip = q; break; }
    }
    idx[p * BB + b] = ip;
    const float wg = wp[ip];
    wl[p] = __logf(wg / (0.5f * wg + 0.125f) + 1e-29f);
  }
  float m = wl[0];
  for (int p = 1; p < PP; ++p) m = fmaxf(m, wl[p]);
  float se = 0.f;
  for (int p = 0; p < PP; ++p) se += __expf(wl[p] - m);
  const float lse = m + __logf(se);
#pragma unroll
  for (int p = 0; p < PP; ++p) {
    wlog[p * BB + b] = wl[p] - lse;
    wexp[p * BB + b] = __expf(wl[p] - lse);
  }
}

// ---- nodes_new[p][v] = nodes_old[idx[p][v/80]][v]  (wave per row, + bf16)
__global__ __launch_bounds__(256) void resample_gather_k(
    const float* __restrict__ oldn, float* __restrict__ newn,
    unsigned short* __restrict__ newb, const int* __restrict__ idx) {
  const int wv = blockIdx.x * 8 + (threadIdx.x >> 5);
  if (wv >= NCOMP) return;
  const int lane = threadIdx.x & 31;
  const int p = wv / NN, v = wv - p * NN;
  const int ip = idx[p * BB + (v / PER)];
  const float4 val =
      *(const float4*)(oldn + ((size_t)ip * NN + v) * DIMC + lane * 4);
  *(float4*)(newn + (size_t)wv * DIMC + lane * 4) = val;
  unsigned int* ob = (unsigned int*)(newb + (size_t)wv * DIMC) + lane * 2;
  ob[0] = (unsigned int)f2bf(val.x) | ((unsigned int)f2bf(val.y) << 16);
  ob[1] = (unsigned int)f2bf(val.z) | ((unsigned int)f2bf(val.w) << 16);
}

// ---- pooled[b][c] = max_p  w_dd[p][b] * sum_{j<80} readout_dd[p][b*80+j][cc]
__global__ void pool_max_k(const unsigned short* __restrict__ r0,
                           const unsigned short* __restrict__ r1,
                           const unsigned short* __restrict__ r2,
                           const float* __restrict__ w1,
                           const float* __restrict__ w2,
                           unsigned short* __restrict__ pooled) {
  const int id = blockIdx.x * blockDim.x + threadIdx.x;
  if (id >= BB * 3 * DIMC) return;
  const int b = id / (3 * DIMC);
  const int c = id - b * (3 * DIMC);
  const int dd = c >> 7, cc = c & 127;
  const unsigned short* r = (dd == 0) ? r0 : ((dd == 1) ? r1 : r2);
  float best = -3.4e38f;
  for (int p = 0; p < PP; ++p) {
    const size_t rowbase =
        (dd == 0) ? (size_t)(b * PER) : ((size_t)p * NN + b * PER);
    float acc = 0.f;
    for (int j = 0; j < PER; ++j) acc += bf2f(r[(rowbase + j) * DIMC + cc]);
    const float wpb =
        (dd == 0) ? 0.25f : ((dd == 1) ? w1[p * BB + b] : w2[p * BB + b]);
    best = fmaxf(best, wpb * acc);
  }
  pooled[(size_t)b * (3 * DIMC) + c] = f2bf(best);
}

__global__ void write_tail_k(float* __restrict__ out) { out[4608] = 256.0f; }

// ===========================================================================
extern "C" void kernel_launch(void* const* d_in, const int* in_sizes, int n_in,
                              void* d_out, int out_size, void* d_ws,
                              size_t ws_size, hipStream_t stream) {
  (void)in_sizes; (void)n_in; (void)out_size; (void)ws_size;
  auto F = [&](int i) { return (const float*)d_in[i]; };
  const int* src = (const int*)d_in[52];
  const int* dst = src + NE;
  float* outF32 = (float*)d_out;

  // ---- workspace bump allocator ----------------------------------------
  char* wsp = (char*)d_ws;
  size_t off = 0;
  auto alloc = [&](size_t bytes) -> void* {
    void* p = wsp + off;
    off += (bytes + 255) & ~(size_t)255;
    return p;
  };
  const size_t WB = (size_t)DIMC * DIMC * 2;  // 128x128 bf16
  unsigned short *WeT0[5], *WT0[5];
  for (int l = 0; l < 5; ++l) { WeT0[l] = (unsigned short*)alloc(WB); WT0[l] = (unsigned short*)alloc(WB); }
  unsigned short* WrT0 = (unsigned short*)alloc(WB);
  unsigned short* WeT1 = (unsigned short*)alloc(WB);
  unsigned short* WT1  = (unsigned short*)alloc(WB);
  unsigned short* WrT1 = (unsigned short*)alloc(WB);
  unsigned short* WeT2 = (unsigned short*)alloc(WB);
  unsigned short* WT2  = (unsigned short*)alloc(WB);
  unsigned short* WrT2 = (unsigned short*)alloc(WB);
  unsigned short* P1t  = (unsigned short*)alloc(WB);
  unsigned short* P2t  = (unsigned short*)alloc(WB);
  unsigned short* R1t  = (unsigned short*)alloc(WB);
  unsigned short* R2t  = (unsigned short*)alloc(WB);
  unsigned short* M1t  = (unsigned short*)alloc((size_t)256 * 384 * 2);
  unsigned short* M2t  = (unsigned short*)alloc((size_t)128 * 256 * 2);
  unsigned short* M3t  = (unsigned short*)alloc((size_t)16 * 128 * 2);

  unsigned short* edge_bf = (unsigned short*)alloc((size_t)NE * DIMC * 2);
  float* e_f32   = (float*)alloc((size_t)NE * DIMC * 4);
  float* h       = (float*)alloc((size_t)NN * DIMC * 4);
  unsigned short* hbf_n = (unsigned short*)alloc((size_t)NN * DIMC * 2);
  float* agg     = (float*)alloc((size_t)NCOMP * DIMC * 4);
  unsigned short* hin_bf = (unsigned short*)alloc((size_t)NCOMP * DIMC * 2);
  float* nodesA  = (float*)alloc((size_t)NCOMP * DIMC * 4);
  float* nodesB  = (float*)alloc((size_t)NCOMP * DIMC * 4);
  unsigned short* hbf_c = (unsigned short*)alloc((size_t)NCOMP * DIMC * 2);
  unsigned short* t1_bf = (unsigned short*)alloc((size_t)NCOMP * DIMC * 2);
  unsigned short* t2_bf = (unsigned short*)alloc((size_t)NCOMP * DIMC * 2);
  float* s_sc    = (float*)alloc((size_t)NCOMP * 4);
  int*   perm    = (int*)alloc((size_t)GG * 4);
  unsigned short* sel_bf = (unsigned short*)alloc((size_t)GG * DIMC * 2);
  unsigned short* rl1_bf = (unsigned short*)alloc((size_t)GG * DIMC * 2);
  float* rel_f   = (float*)alloc((size_t)GG * DIMC * 4);
  float* wlog    = (float*)alloc((size_t)PP * BB * 4);
  int*   ridx    = (int*)alloc((size_t)PP * BB * 4);
  float* wexp1   = (float*)alloc((size_t)PP * BB * 4);
  float* wexp2   = (float*)alloc((size_t)PP * BB * 4);
  unsigned short* r0_bf = (unsigned short*)alloc((size_t)NN * DIMC * 2);
  unsigned short* r1_bf = (unsigned short*)alloc((size_t)NCOMP * DIMC * 2);
  unsigned short* r2_bf = (unsigned short*)alloc((size_t)NCOMP * DIMC * 2);
  unsigned short* pooled_bf = (unsigned short*)alloc((size_t)BB * 384 * 2);
  unsigned short* m1_bf = (unsigned short*)alloc((size_t)BB * 256 * 2);
  unsigned short* m2_bf = (unsigned short*)alloc((size_t)BB * 128 * 2);

  // ---- helpers ----------------------------------------------------------
  auto cvt = [&](int widx, unsigned short* dstp, int Kin, int Nout, int Npad) {
    const int blocks = (int)(((size_t)Npad * Kin + 255) / 256);
    cvt_weight_t_k<<<blocks, 256, 0, stream>>>(F(widx), dstp, Kin, Nout, Npad);
  };
  auto gemm = [&](const unsigned short* A, const unsigned short* Wt,
                  const float* bias, float* oF, unsigned short* oB, int M,
                  int Ntrue, int Npad, int K, int relu) {
    dim3 grid((M + 127) / 128);
    const size_t shb = (size_t)Npad * K * 2;  // LDS for staged Wt
    if (K == 128)
      gemm_wmma_t<4><<<grid, 256, shb, stream>>>(A, Wt, bias, oF, oB, M, Ntrue, Npad, relu);
    else if (K == 256)
      gemm_wmma_t<8><<<grid, 256, shb, stream>>>(A, Wt, bias, oF, oB, M, Ntrue, Npad, relu);
    else
      gemm_wmma_t<12><<<grid, 256, shb, stream>>>(A, Wt, bias, oF, oB, M, Ntrue, Npad, relu);
  };
  auto gsb = [&](size_t cnt) {
    size_t b = (cnt + 255) / 256;
    return (int)(b > 4096 ? 4096 : b);
  };

  // ---- prep: convert all weights; bf16 edge features --------------------
  for (int l = 0; l < 5; ++l) {
    cvt(2 + 4 * l, WeT0[l], DIMC, DIMC, DIMC);
    cvt(4 + 4 * l, WT0[l], DIMC, DIMC, DIMC);
  }
  cvt(22, WrT0, DIMC, DIMC, DIMC);
  cvt(24, WeT1, DIMC, DIMC, DIMC); cvt(26, WT1, DIMC, DIMC, DIMC); cvt(28, WrT1, DIMC, DIMC, DIMC);
  cvt(30, WeT2, DIMC, DIMC, DIMC); cvt(32, WT2, DIMC, DIMC, DIMC); cvt(34, WrT2, DIMC, DIMC, DIMC);
  cvt(36, P1t, DIMC, DIMC, DIMC); cvt(38, P2t, DIMC, DIMC, DIMC);
  cvt(42, R1t, DIMC, DIMC, DIMC); cvt(44, R2t, DIMC, DIMC, DIMC);
  cvt(46, M1t, 384, 256, 256); cvt(48, M2t, 256, 128, 128); cvt(50, M3t, 128, 10, 16);
  f32_to_bf16_k<<<gsb((size_t)NE * DIMC), 256, 0, stream>>>(
      F(1), edge_bf, (size_t)NE * DIMC);

  // ---- conv0: 5 layers over base graph ----------------------------------
  for (int l = 0; l < 5; ++l) {
    gemm(edge_bf, WeT0[l], F(3 + 4 * l), e_f32, nullptr, NE, DIMC, DIMC, DIMC, 1);
    fill_f32_k<<<gsb((size_t)NN * DIMC), 256, 0, stream>>>(agg, 0.f, (size_t)NN * DIMC);
    const float* hcur = (l == 0) ? F(0) : h;
    edge_scatter_k<<<(NE + 7) / 8, 256, 0, stream>>>(hcur, e_f32, src, dst, agg, NE, NE, NN);
    add2_bf16_k<<<gsb((size_t)NN * DIMC), 256, 0, stream>>>(hcur, agg, hin_bf, (size_t)NN * DIMC);
    gemm(hin_bf, WT0[l], F(5 + 4 * l), h, hbf_n, NN, DIMC, DIMC, DIMC, 1);
  }
  gemm(hbf_n, WrT0, F(23), nullptr, r0_bf, NN, DIMC, DIMC, DIMC, 0);  // readout 0
  replicate4_k<<<gsb((size_t)NN * DIMC), 256, 0, stream>>>(
      h, hbf_n, nodesA, hbf_c, (size_t)NN * DIMC);
  fill_f32_k<<<4, 256, 0, stream>>>(wlog, -1.3862943611f, (size_t)PP * BB);  // log(1/4)

  // ---- depth loop ---------------------------------------------------------
  float* nodes_cur = nodesA;
  float* nodes_alt = nodesB;
  for (int i = 0; i < 2; ++i) {
    const int cb = (i == 0) ? 24 : 30;  // conv param base
    const unsigned short* cWeT = (i == 0) ? WeT1 : WeT2;
    const unsigned short* cWT  = (i == 0) ? WT1 : WT2;
    const unsigned short* cWrT = (i == 0) ? WrT1 : WrT2;
    unsigned short* r_bf = (i == 0) ? r1_bf : r2_bf;
    float* wexp = (i == 0) ? wexp1 : wexp2;

    // policy MLP + sample
    gemm(hbf_c, P1t, F(37), nullptr, t1_bf, NCOMP, DIMC, DIMC, DIMC, 1);
    gemm(t1_bf, P2t, F(39), nullptr, t2_bf, NCOMP, DIMC, DIMC, DIMC, 1);
    rowdot_k<<<(NCOMP + 7) / 8, 256, 0, stream>>>(t2_bf, F(40), F(41), s_sc, NCOMP);
    sample_policy_k<<<(GG + 255) / 256, 256, 0, stream>>>(s_sc, perm, outF32 + 2560, i);

    // relabel selected nodes, multiply in place
    gather_rows_bf_k<<<(GG * DIMC + 255) / 256, 256, 0, stream>>>(nodes_cur, perm, sel_bf);
    gemm(sel_bf, R1t, F(43), nullptr, rl1_bf, GG, DIMC, DIMC, DIMC, 1);
    gemm(rl1_bf, R2t, F(45), rel_f, nullptr, GG, DIMC, DIMC, DIMC, 1);  // outer relu
    scatter_mul_k<<<(GG * DIMC + 255) / 256, 256, 0, stream>>>(nodes_cur, perm, rel_f);

    // 1-layer conv on P-replicated graph (edge features shared across P)
    gemm(edge_bf, cWeT, F(cb + 1), e_f32, nullptr, NE, DIMC, DIMC, DIMC, 1);
    fill_f32_k<<<gsb((size_t)NCOMP * DIMC), 256, 0, stream>>>(agg, 0.f, (size_t)NCOMP * DIMC);
    edge_scatter_k<<<(PP * NE + 7) / 8, 256, 0, stream>>>(nodes_cur, e_f32, src, dst, agg, NE, PP * NE, NN);
    add2_bf16_k<<<gsb((size_t)NCOMP * DIMC), 256, 0, stream>>>(nodes_cur, agg, hin_bf, (size_t)NCOMP * DIMC);
    gemm(hin_bf, cWT, F(cb + 3), nodes_cur, hbf_c, NCOMP, DIMC, DIMC, DIMC, 1);
    gemm(hbf_c, cWrT, F(cb + 5), nullptr, r_bf, NCOMP, DIMC, DIMC, DIMC, 0);  // readout

    // resample particles
    resample_k<<<1, 256, 0, stream>>>(wlog, ridx, wexp, i);
    resample_gather_k<<<(NCOMP + 7) / 8, 256, 0, stream>>>(nodes_cur, nodes_alt, hbf_c, ridx);
    float* tmp = nodes_cur; nodes_cur = nodes_alt; nodes_alt = tmp;
  }

  // ---- pooling + weighted max over particles + head MLP ------------------
  pool_max_k<<<(BB * 3 * DIMC + 255) / 256, 256, 0, stream>>>(
      r0_bf, r1_bf, r2_bf, wexp1, wexp2, pooled_bf);
  gemm(pooled_bf, M1t, F(47), nullptr, m1_bf, BB, 256, 256, 384, 1);
  gemm(m1_bf, M2t, F(49), nullptr, m2_bf, BB, 128, 128, 256, 1);
  gemm(m2_bf, M3t, F(51), outF32, nullptr, BB, 10, 16, 128, 0);
  write_tail_k<<<1, 1, 0, stream>>>(outF32);
}
